// JointAttention_55396488183867
// MI455X (gfx1250) — matrix operand order repaired and checked by reading, not compiled, not run on hardware
//
#include <hip/hip_runtime.h>
#include <math.h>

// ---------------- CDNA5 WMMA types ----------------
typedef __bf16 bf16x16 __attribute__((ext_vector_type(16)));
typedef __bf16 bf16x8  __attribute__((ext_vector_type(8)));
typedef float  floatx8 __attribute__((ext_vector_type(8)));

static constexpr int DIM  = 1024;
static constexpr int H    = 16;
static constexpr int KVH  = 4;
static constexpr int DH   = 64;
static constexpr int NSEQ = 2048;   // per stream
static constexpr int NTOT = 4096;   // joint sequence

__device__ __forceinline__ floatx8 wmma_bf16(bf16x16 a, bf16x16 b, floatx8 c) {
  return __builtin_amdgcn_wmma_f32_16x16x32_bf16(false, a, false, b, (short)0, c,
                                                 false, false);
}

__device__ __forceinline__ bf16x16 join8(bf16x8 lo, bf16x8 hi) {
  return __builtin_shufflevector(lo, hi, 0, 1, 2, 3, 4, 5, 6, 7,
                                 8, 9, 10, 11, 12, 13, 14, 15);
}

// A fragment: 16x32 bf16 tile, row-major source (origin applied), 16B loads.
// lane L: row m = L&15, half = L>>4; element e -> k = 16*(e>>3) + (e&7) + 8*half
__device__ __forceinline__ bf16x16 load_a_frag(const __bf16* __restrict__ A, int lda) {
  int lane = threadIdx.x & 31;
  int m = lane & 15, half = lane >> 4;
  const __bf16* rp = A + (size_t)m * lda;
  bf16x8 lo = *(const bf16x8*)(rp + 8 * half);        // k = 8*half .. +7
  bf16x8 hi = *(const bf16x8*)(rp + 16 + 8 * half);   // k = 16+8*half .. +7
  return join8(lo, hi);
}

// B fragment from TRANSPOSED source: row-major [N x K] (row n holds its k-run).
// lane L: col n = L&15, half = L>>4; element e -> k = 16*(e>>3) + (e&7) + 8*half
__device__ __forceinline__ bf16x16 load_b_frag_T(const __bf16* __restrict__ Bt,
                                                 int ldt) {
  int lane = threadIdx.x & 31;
  int n = lane & 15, half = lane >> 4;
  const __bf16* rp = Bt + (size_t)n * ldt;
  bf16x8 lo = *(const bf16x8*)(rp + 8 * half);
  bf16x8 hi = *(const bf16x8*)(rp + 16 + 8 * half);
  return join8(lo, hi);
}

// ---------------- elementwise / transpose kernels ----------------

// f32 [K x N] -> bf16 [N x K], LDS-tiled so both global accesses are coalesced.
__global__ void transpose_cvt(const float* __restrict__ in, __bf16* __restrict__ out,
                              int K, int N) {
  __shared__ float tile[16][17];
  int n0 = blockIdx.x * 16, k0 = blockIdx.y * 16;
  int tx = threadIdx.x, ty = threadIdx.y;
  tile[ty][tx] = in[(size_t)(k0 + ty) * N + n0 + tx];
  __syncthreads();
  out[(size_t)(n0 + ty) * K + k0 + tx] = (__bf16)tile[tx][ty];
}

__global__ void rmsnorm_bf16(const float* __restrict__ x, const float* __restrict__ g,
                             __bf16* __restrict__ out) {
  int row = blockIdx.x;
  const float* xr = x + (size_t)row * DIM;
  __shared__ float red[256];
  float s = 0.f;
  for (int i = threadIdx.x; i < DIM; i += 256) { float v = xr[i]; s += v * v; }
  red[threadIdx.x] = s;
  __syncthreads();
  for (int off = 128; off > 0; off >>= 1) {
    if ((int)threadIdx.x < off) red[threadIdx.x] += red[threadIdx.x + off];
    __syncthreads();
  }
  float scale = rsqrtf(red[0] / (float)DIM + 1e-6f);
  for (int i = threadIdx.x; i < DIM; i += 256)
    out[(size_t)row * DIM + i] = (__bf16)(xr[i] * scale * g[i]);
}

// RoPE on Q projections: src [NSEQ x (H*DH)] -> Qj [H][NTOT][DH] at posOff.
__global__ void rope_q(const __bf16* __restrict__ Q, __bf16* __restrict__ Qj, int posOff) {
  int idx = blockIdx.x * blockDim.x + threadIdx.x;
  int i = idx & 31;
  int h = (idx >> 5) & 15;
  int p = idx >> 9;
  if (p >= NSEQ) return;
  float inv = powf(10000.f, -((float)(2 * i)) / 64.f);
  float ang = (float)(2 * p) * inv;           // t = p * (4096/2048)
  float c = cosf(ang), s = sinf(ang);
  const __bf16* src = Q + (size_t)p * (H * DH) + h * DH;
  float x1 = (float)src[i];
  float x2 = (float)src[i + 32];
  __bf16* dst = Qj + ((size_t)h * NTOT + posOff + p) * DH;
  dst[i]      = (__bf16)(x1 * c - x2 * s);
  dst[i + 32] = (__bf16)(x2 * c + x1 * s);
}

// KV: src [NSEQ x (2*KVH*DH)] (K cols 0..255, V cols 256..511).
// RoPE K into Kj [KVH][NTOT][DH]; V into TRANSPOSED Vj [KVH][DH][NTOT].
__global__ void rope_kv(const __bf16* __restrict__ KV, __bf16* __restrict__ Kj,
                        __bf16* __restrict__ Vj, int posOff) {
  int idx = blockIdx.x * blockDim.x + threadIdx.x;
  int i = idx & 31;
  int h = (idx >> 5) & 3;
  int p = idx >> 7;
  if (p >= NSEQ) return;
  float inv = powf(10000.f, -((float)(2 * i)) / 64.f);
  float ang = (float)(2 * p) * inv;
  float c = cosf(ang), s = sinf(ang);
  const __bf16* kr = KV + (size_t)p * (2 * KVH * DH) + h * DH;
  const __bf16* vr = kr + KVH * DH;
  __bf16* kd = Kj + ((size_t)h * NTOT + posOff + p) * DH;
  kd[i]      = (__bf16)((float)kr[i] * c - (float)kr[i + 32] * s);
  kd[i + 32] = (__bf16)((float)kr[i + 32] * c + (float)kr[i] * s);
  __bf16* vd = Vj + (size_t)h * DH * NTOT;
  vd[(size_t)i * NTOT + posOff + p]        = vr[i];
  vd[(size_t)(i + 32) * NTOT + posOff + p] = vr[i + 32];
}

// ---------------- WMMA GEMMs (B pre-transposed: Bt is [N x K]) ----------------
// One wave computes a 32(M) x 64(N) tile: 2 A-frags x 4 B-frags -> 8 WMMA / k-step.
__global__ void __launch_bounds__(32)
gemm_bf16_bf16(const __bf16* __restrict__ A, const __bf16* __restrict__ Bt,
               __bf16* __restrict__ C, int M, int N, int K) {
  int col0 = blockIdx.x * 64;
  int row0 = blockIdx.y * 32;
  int lane = threadIdx.x & 31;
  int n = lane & 15, half = lane >> 4;
  floatx8 acc[2][4] = {};
  for (int k0 = 0; k0 < K; k0 += 32) {
    if (k0 + 32 < K)
      __builtin_prefetch(A + (size_t)(row0 + (lane & 15)) * K + k0 + 32, 0, 1);
    bf16x16 a0 = load_a_frag(A + (size_t)row0 * K + k0, K);
    bf16x16 a1 = load_a_frag(A + (size_t)(row0 + 16) * K + k0, K);
#pragma unroll
    for (int j = 0; j < 4; ++j) {
      bf16x16 bf = load_b_frag_T(Bt + (size_t)(col0 + 16 * j) * K + k0, K);
      acc[0][j] = wmma_bf16(a0, bf, acc[0][j]);
      acc[1][j] = wmma_bf16(a1, bf, acc[1][j]);
    }
  }
#pragma unroll
  for (int r = 0; r < 2; ++r)
#pragma unroll
    for (int j = 0; j < 4; ++j)
#pragma unroll
      for (int e = 0; e < 8; ++e)
        C[(size_t)(row0 + 16 * r + e + 8 * half) * N + col0 + 16 * j + n] =
            (__bf16)acc[r][j][e];
}

// Same tiling, f32 output + bias.
__global__ void __launch_bounds__(32)
gemm_bf16_f32_bias(const __bf16* __restrict__ A, const __bf16* __restrict__ Bt,
                   const float* __restrict__ bias, float* __restrict__ C,
                   int M, int N, int K) {
  int col0 = blockIdx.x * 64;
  int row0 = blockIdx.y * 32;
  int lane = threadIdx.x & 31;
  int n = lane & 15, half = lane >> 4;
  floatx8 acc[2][4] = {};
  for (int k0 = 0; k0 < K; k0 += 32) {
    if (k0 + 32 < K)
      __builtin_prefetch(A + (size_t)(row0 + (lane & 15)) * K + k0 + 32, 0, 1);
    bf16x16 a0 = load_a_frag(A + (size_t)row0 * K + k0, K);
    bf16x16 a1 = load_a_frag(A + (size_t)(row0 + 16) * K + k0, K);
#pragma unroll
    for (int j = 0; j < 4; ++j) {
      bf16x16 bf = load_b_frag_T(Bt + (size_t)(col0 + 16 * j) * K + k0, K);
      acc[0][j] = wmma_bf16(a0, bf, acc[0][j]);
      acc[1][j] = wmma_bf16(a1, bf, acc[1][j]);
    }
  }
#pragma unroll
  for (int r = 0; r < 2; ++r)
#pragma unroll
    for (int j = 0; j < 4; ++j) {
      float b = bias[col0 + 16 * j + n];
#pragma unroll
      for (int e = 0; e < 8; ++e)
        C[(size_t)(row0 + 16 * r + e + 8 * half) * N + col0 + 16 * j + n] =
            acc[r][j][e] + b;
    }
}

// ---------------- Flash attention over the joint sequence ----------------
// Qj [H][NTOT][DH], Kj [KVH][NTOT][DH], Vj [KVH][DH][NTOT] (transposed).
// Scores get the reference's combined factor scale^2 = 1/64 in f32.
// Output: O [NTOT][H*DH] bf16 (merged-head layout, rows 0..2047 = stream a).
__global__ void __launch_bounds__(32)
flash_attn(const __bf16* __restrict__ Qj, const __bf16* __restrict__ Kj,
           const __bf16* __restrict__ Vj, __bf16* __restrict__ O) {
  int h  = blockIdx.y;
  int q0 = blockIdx.x * 16;
  int kvh = h & (KVH - 1);                 // jnp.tile => head h uses kv head h%4
  int lane = threadIdx.x & 31;
  int nn = lane & 15, half = lane >> 4;

  const __bf16* Qbase = Qj + ((size_t)h * NTOT + q0) * DH;
  bf16x16 aq0 = load_a_frag(Qbase, DH);        // d 0..31
  bf16x16 aq1 = load_a_frag(Qbase + 32, DH);   // d 32..63

  const __bf16* Kb = Kj + (size_t)kvh * NTOT * DH;
  const __bf16* Vb = Vj + (size_t)kvh * DH * NTOT;

  floatx8 o0 = {}, o1 = {}, o2 = {}, o3 = {};
  float mrow[8], lrow[8];
#pragma unroll
  for (int e = 0; e < 8; ++e) { mrow[e] = -3.0e38f; lrow[e] = 0.f; }

  __shared__ __bf16 Plds[16 * 32];
  const float sc = 1.0f / 64.0f;

  for (int kb = 0; kb < NTOT; kb += 32) {
    // ---- scores: S[16q x 32k] as two 16x16 D-tiles ----
    floatx8 s0 = {}, s1 = {};
    {
      // lane n = key index; per-lane contiguous along d
      bf16x16 b0 = load_b_frag_T(Kb + (size_t)kb * DH, DH);
      bf16x16 b1 = load_b_frag_T(Kb + (size_t)kb * DH + 32, DH);
      s0 = wmma_bf16(aq0, b0, s0);
      s0 = wmma_bf16(aq1, b1, s0);
    }
    {
      bf16x16 b0 = load_b_frag_T(Kb + (size_t)(kb + 16) * DH, DH);
      bf16x16 b1 = load_b_frag_T(Kb + (size_t)(kb + 16) * DH + 32, DH);
      s1 = wmma_bf16(aq0, b0, s1);
      s1 = wmma_bf16(aq1, b1, s1);
    }
    // ---- online softmax (row m = e + 8*half lives in one 16-lane half) ----
    float p0[8], p1[8], alpha[8];
#pragma unroll
    for (int e = 0; e < 8; ++e) {
      float v0 = s0[e] * sc, v1 = s1[e] * sc;
      float rm = fmaxf(v0, v1);
#pragma unroll
      for (int msk = 1; msk < 16; msk <<= 1)
        rm = fmaxf(rm, __shfl_xor(rm, msk, 32));
      float mn = fmaxf(mrow[e], rm);
      float al = __expf(mrow[e] - mn);
      float e0 = __expf(v0 - mn), e1 = __expf(v1 - mn);
      float rs = e0 + e1;
#pragma unroll
      for (int msk = 1; msk < 16; msk <<= 1)
        rs += __shfl_xor(rs, msk, 32);
      lrow[e] = lrow[e] * al + rs;
      mrow[e] = mn;
      alpha[e] = al; p0[e] = e0; p1[e] = e1;
    }
#pragma unroll
    for (int e = 0; e < 8; ++e) {
      o0[e] *= alpha[e]; o1[e] *= alpha[e]; o2[e] *= alpha[e]; o3[e] *= alpha[e];
    }
    // ---- transpose P: D-layout -> LDS -> A-layout (two ds_load_b128) ----
#pragma unroll
    for (int e = 0; e < 8; ++e) {
      Plds[(e + 8 * half) * 32 + nn]      = (__bf16)p0[e];
      Plds[(e + 8 * half) * 32 + 16 + nn] = (__bf16)p1[e];
    }
    __syncthreads();
    bf16x16 ap;
    {
      const __bf16* rp = Plds + nn * 32;   // lane's query row m = nn
      bf16x8 lo = *(const bf16x8*)(rp + 8 * half);
      bf16x8 hi = *(const bf16x8*)(rp + 16 + 8 * half);
      ap = join8(lo, hi);
    }
    __syncthreads();
    // ---- O += P x V  (V transposed: lane n = dim, contiguous along keys) ----
    {
      const __bf16* Vt = Vb + kb;
      bf16x16 bv0 = load_b_frag_T(Vt + (size_t)0  * NTOT, NTOT);
      bf16x16 bv1 = load_b_frag_T(Vt + (size_t)16 * NTOT, NTOT);
      bf16x16 bv2 = load_b_frag_T(Vt + (size_t)32 * NTOT, NTOT);
      bf16x16 bv3 = load_b_frag_T(Vt + (size_t)48 * NTOT, NTOT);
      o0 = wmma_bf16(ap, bv0, o0);
      o1 = wmma_bf16(ap, bv1, o1);
      o2 = wmma_bf16(ap, bv2, o2);
      o3 = wmma_bf16(ap, bv3, o3);
    }
  }
#pragma unroll
  for (int e = 0; e < 8; ++e) {
    float inv_l = 1.0f / lrow[e];
    int row = q0 + e + 8 * half;
    __bf16* orow = O + (size_t)row * (H * DH) + h * DH;
    orow[nn]      = (__bf16)(o0[e] * inv_l);
    orow[16 + nn] = (__bf16)(o1[e] * inv_l);
    orow[32 + nn] = (__bf16)(o2[e] * inv_l);
    orow[48 + nn] = (__bf16)(o3[e] * inv_l);
  }
}

// ---------------- host launch ----------------
extern "C" void kernel_launch(void* const* d_in, const int* in_sizes, int n_in,
                              void* d_out, int out_size, void* d_ws, size_t ws_size,
                              hipStream_t stream) {
  const float* x      = (const float*)d_in[0];
  const float* a      = (const float*)d_in[1];
  const float* g_x    = (const float*)d_in[2];
  const float* g_a    = (const float*)d_in[3];
  const float* Wq_x   = (const float*)d_in[4];
  const float* Wkv_x  = (const float*)d_in[5];
  const float* Wq_a   = (const float*)d_in[6];
  const float* Wkv_a  = (const float*)d_in[7];
  const float* Wout_x = (const float*)d_in[8];
  const float* bout_x = (const float*)d_in[9];
  const float* Wout_a = (const float*)d_in[10];
  const float* bout_a = (const float*)d_in[11];
  float* out = (float*)d_out;

  // workspace carve-out (bf16 buffers), 256B aligned
  char* ws = (char*)d_ws;
  size_t off = 0;
  auto alloc = [&](size_t bytes) -> void* {
    void* p = ws + off;
    off += (bytes + 255) & ~(size_t)255;
    return p;
  };
  // transposed bf16 weights: [N x K]
  __bf16* wQxT  = (__bf16*)alloc((size_t)(H * DH) * DIM * 2);
  __bf16* wKVxT = (__bf16*)alloc((size_t)(2 * KVH * DH) * DIM * 2);
  __bf16* wQaT  = (__bf16*)alloc((size_t)(H * DH) * DIM * 2);
  __bf16* wKVaT = (__bf16*)alloc((size_t)(2 * KVH * DH) * DIM * 2);
  __bf16* wOxT  = (__bf16*)alloc((size_t)DIM * (H * DH) * 2);
  __bf16* wOaT  = (__bf16*)alloc((size_t)DIM * (H * DH) * 2);
  __bf16* xn    = (__bf16*)alloc((size_t)NSEQ * DIM * 2);
  __bf16* an    = (__bf16*)alloc((size_t)NSEQ * DIM * 2);
  __bf16* Qx    = (__bf16*)alloc((size_t)NSEQ * H * DH * 2);
  __bf16* Qa    = (__bf16*)alloc((size_t)NSEQ * H * DH * 2);
  __bf16* KVx   = (__bf16*)alloc((size_t)NSEQ * 2 * KVH * DH * 2);
  __bf16* KVa   = (__bf16*)alloc((size_t)NSEQ * 2 * KVH * DH * 2);
  __bf16* Qj    = (__bf16*)alloc((size_t)H * NTOT * DH * 2);
  __bf16* Kj    = (__bf16*)alloc((size_t)KVH * NTOT * DH * 2);
  __bf16* Vj    = (__bf16*)alloc((size_t)KVH * DH * NTOT * 2);   // transposed
  __bf16* attn  = (__bf16*)alloc((size_t)NTOT * H * DH * 2);
  (void)ws_size; (void)in_sizes; (void)n_in; (void)out_size;

  // 1) weights -> bf16, transposed to [N x K]
  auto tcvt = [&](const float* src, __bf16* dst, int K, int N) {
    transpose_cvt<<<dim3(N / 16, K / 16), dim3(16, 16), 0, stream>>>(src, dst, K, N);
  };
  tcvt(Wq_x,  wQxT,  DIM, H * DH);
  tcvt(Wkv_x, wKVxT, DIM, 2 * KVH * DH);
  tcvt(Wq_a,  wQaT,  DIM, H * DH);
  tcvt(Wkv_a, wKVaT, DIM, 2 * KVH * DH);
  tcvt(Wout_x, wOxT, H * DH, DIM);
  tcvt(Wout_a, wOaT, H * DH, DIM);

  // 2) rmsnorm
  rmsnorm_bf16<<<dim3(NSEQ), dim3(256), 0, stream>>>(x, g_x, xn);
  rmsnorm_bf16<<<dim3(NSEQ), dim3(256), 0, stream>>>(a, g_a, an);

  // 3) QKV projections (WMMA, 32x64 tiles per wave)
  gemm_bf16_bf16<<<dim3((H * DH) / 64, NSEQ / 32), dim3(32), 0, stream>>>(
      xn, wQxT, Qx, NSEQ, H * DH, DIM);
  gemm_bf16_bf16<<<dim3((2 * KVH * DH) / 64, NSEQ / 32), dim3(32), 0, stream>>>(
      xn, wKVxT, KVx, NSEQ, 2 * KVH * DH, DIM);
  gemm_bf16_bf16<<<dim3((H * DH) / 64, NSEQ / 32), dim3(32), 0, stream>>>(
      an, wQaT, Qa, NSEQ, H * DH, DIM);
  gemm_bf16_bf16<<<dim3((2 * KVH * DH) / 64, NSEQ / 32), dim3(32), 0, stream>>>(
      an, wKVaT, KVa, NSEQ, 2 * KVH * DH, DIM);

  // 4) RoPE into joint head-major buffers (stream a at rows 0.., x at 2048..)
  rope_q<<<dim3((NSEQ * H * 32) / 256), dim3(256), 0, stream>>>(Qa, Qj, 0);
  rope_q<<<dim3((NSEQ * H * 32) / 256), dim3(256), 0, stream>>>(Qx, Qj, NSEQ);
  rope_kv<<<dim3((NSEQ * KVH * 32) / 256), dim3(256), 0, stream>>>(KVa, Kj, Vj, 0);
  rope_kv<<<dim3((NSEQ * KVH * 32) / 256), dim3(256), 0, stream>>>(KVx, Kj, Vj, NSEQ);

  // 5) joint flash attention (WMMA + LDS transpose)
  flash_attn<<<dim3(NTOT / 16, H), dim3(32), 0, stream>>>(Qj, Kj, Vj, attn);

  // 6) output projections (+bias) straight into d_out: out_x first, then out_a
  gemm_bf16_f32_bias<<<dim3(DIM / 64, NSEQ / 32), dim3(32), 0, stream>>>(
      attn + (size_t)NSEQ * H * DH, wOxT, bout_x, out, NSEQ, DIM, H * DH);
  gemm_bf16_f32_bias<<<dim3(DIM / 64, NSEQ / 32), dim3(32), 0, stream>>>(
      attn, wOaT, bout_a, out + (size_t)NSEQ * DIM, NSEQ, DIM, H * DH);
}
